// Transformer_45011257262745
// MI455X (gfx1250) — matrix operand order repaired and checked by reading, not compiled, not run on hardware
//
#include <hip/hip_runtime.h>
#include <hip/hip_bf16.h>

#define D_MODEL 1024
#define N_HEAD  16
#define DHEAD   64
#define D_FF    4096
#define VOCAB   32000
#define BATCH   4
#define SEQ     512
#define ROWS    (BATCH * SEQ)   // 2048

typedef __attribute__((ext_vector_type(16))) __bf16          v16bf;
typedef __attribute__((ext_vector_type(8)))  float           v8f;
typedef __attribute__((ext_vector_type(8)))  unsigned short  u16x8;

union FragAB {
  v16bf          bf;
  u16x8          u[2];
  unsigned short s[16];
};
static_assert(sizeof(v16bf) == 32, "bf16x16 must be 32 bytes");

__device__ __forceinline__ unsigned short f2bf_bits(float f) {
  unsigned u = __float_as_uint(f);
  u += 0x7FFFu + ((u >> 16) & 1u);      // round-to-nearest-even
  return (unsigned short)(u >> 16);
}

__device__ __forceinline__ void pack8(unsigned short* d, float4 a, float4 b) {
  d[0] = f2bf_bits(a.x); d[1] = f2bf_bits(a.y); d[2] = f2bf_bits(a.z); d[3] = f2bf_bits(a.w);
  d[4] = f2bf_bits(b.x); d[5] = f2bf_bits(b.y); d[6] = f2bf_bits(b.z); d[7] = f2bf_bits(b.w);
}

__device__ __forceinline__ v8f wmma_bf16(const FragAB& a, const FragAB& b, v8f c) {
  return __builtin_amdgcn_wmma_f32_16x16x32_bf16(
      /*neg_a=*/false, a.bf, /*neg_b=*/false, b.bf,
      /*c_mod=*/(short)0, c, /*reuse_a=*/false, /*reuse_b=*/false);
}

// ---------------------------------------------------------------------------
// Embedding gather + positional encoding:  x[b,l,d] = emb[tok[b,l],d] + pe(l,d)
// pe(l,d) = sin/cos( l * 10000^(-2d/D) ), sin on even d, cos on odd d.
// ---------------------------------------------------------------------------
__global__ __launch_bounds__(256)
void embed_pe(const int* __restrict__ tok, const float* __restrict__ emb,
              float* __restrict__ X) {
  int idx = blockIdx.x * 256 + threadIdx.x;      // [0, ROWS*D_MODEL)
  int d   = idx & (D_MODEL - 1);
  int bl  = idx >> 10;
  int l   = bl & (SEQ - 1);
  int t   = tok[bl];
  float angle = (float)l * __powf(10000.0f, -(float)d * (2.0f / (float)D_MODEL));
  float pe = (d & 1) ? __cosf(angle) : __sinf(angle);
  X[idx] = emb[(size_t)t * D_MODEL + d] + pe;
}

// ---------------------------------------------------------------------------
// Tiled GEMM: C[M,N] = A[M,K] @ W[K,N] (+bias) (+relu), bf16 WMMA, f32 accum.
// Block = 256 thr = 8 waves; tile 128x128, K-step 32, software-pipelined:
// tile k+1 global loads are issued during tile k's WMMAs; tile k+2 prefetched.
// Wave (wm in 0..3, wn in 0..1) owns a 32x64 sub-tile = 2x4 WMMA frags.
// ---------------------------------------------------------------------------
#define TM   128
#define TN   128
#define TK   32
#define LPIT 40   // LDS pitch in halves: 80B, multiple of 16B -> aligned b128 runs

__global__ __launch_bounds__(256)
void gemm_bf16_wmma(const float* __restrict__ A, const float* __restrict__ W,
                    const float* __restrict__ bias, float* __restrict__ C,
                    int M, int N, int K, int relu) {
  __shared__ alignas(16) unsigned short As[TM * LPIT];   // [m][k]
  __shared__ alignas(16) unsigned short Bs[TN * LPIT];   // transposed: [n][k]

  const int tid  = threadIdx.x;
  const int lane = tid & 31;
  const int wave = tid >> 5;
  const int wm0  = (wave & 3) * 32;
  const int wn0  = (wave >> 2) * 64;
  const int gm0  = blockIdx.y * TM;
  const int gn0  = blockIdx.x * TN;
  const int lrow = lane & 15;
  const int lhi  = lane >> 4;

  v8f acc[2][4] = {};

  const int arow = tid >> 1;            // 0..127
  const int aks  = (tid & 1) * 16;      // 0 / 16
  const int bkr  = tid >> 3;            // 0..31
  const int bns  = (tid & 7) * 16;      // 0..112

  const float* aBase = A + (size_t)(gm0 + arow) * K + aks;
  const float* bBase = W + (size_t)bkr * N + gn0 + bns;
  const size_t bStep = (size_t)N;   // advance one k-row

  float4 ar[4], br[4];
  // prologue: stage tile 0 into registers
  {
    const float4* sa = (const float4*)(aBase);
    ar[0] = sa[0]; ar[1] = sa[1]; ar[2] = sa[2]; ar[3] = sa[3];
    const float4* sb = (const float4*)(bBase);
    br[0] = sb[0]; br[1] = sb[1]; br[2] = sb[2]; br[3] = sb[3];
  }

  for (int kk = 0; kk < K; kk += TK) {
    // ---- regs -> LDS (bf16) ----
    {
      union { u16x8 v; unsigned short s[8]; } p0, p1;
      pack8(p0.s, ar[0], ar[1]);
      pack8(p1.s, ar[2], ar[3]);
      *(u16x8*)&As[arow * LPIT + aks]     = p0.v;
      *(u16x8*)&As[arow * LPIT + aks + 8] = p1.v;
      unsigned short t[16];
      pack8(t,     br[0], br[1]);
      pack8(t + 8, br[2], br[3]);
      #pragma unroll
      for (int j = 0; j < 16; ++j)
        Bs[(bns + j) * LPIT + bkr] = t[j];
    }
    __syncthreads();

    // ---- issue next tile's global loads (wait deferred to next store phase) ----
    if (kk + TK < K) {
      const float4* sa = (const float4*)(aBase + kk + TK);
      ar[0] = sa[0]; ar[1] = sa[1]; ar[2] = sa[2]; ar[3] = sa[3];
      const float4* sb = (const float4*)(bBase + (size_t)(kk + TK) * bStep);
      br[0] = sb[0]; br[1] = sb[1]; br[2] = sb[2]; br[3] = sb[3];
    }
    // ---- prefetch tile k+2 into cache (global_prefetch_b8) ----
    if (kk + 2 * TK < K) {
      __builtin_prefetch(aBase + kk + 2 * TK, 0, 1);
      __builtin_prefetch(bBase + (size_t)(kk + 2 * TK) * bStep, 0, 1);
    }

    // ---- fragment loads + 8 WMMAs ----
    FragAB a0, a1, bf[4];
    {
      int base0 = (wm0 + lrow) * LPIT + lhi * 8;
      int base1 = (wm0 + 16 + lrow) * LPIT + lhi * 8;
      a0.u[0] = *(const u16x8*)&As[base0];  a0.u[1] = *(const u16x8*)&As[base0 + 16];
      a1.u[0] = *(const u16x8*)&As[base1];  a1.u[1] = *(const u16x8*)&As[base1 + 16];
    }
    #pragma unroll
    for (int f = 0; f < 4; ++f) {
      int base = (wn0 + f * 16 + lrow) * LPIT + lhi * 8;
      bf[f].u[0] = *(const u16x8*)&Bs[base];
      bf[f].u[1] = *(const u16x8*)&Bs[base + 16];
    }
    #pragma unroll
    for (int f = 0; f < 4; ++f) {
      acc[0][f] = wmma_bf16(a0, bf[f], acc[0][f]);
      acc[1][f] = wmma_bf16(a1, bf[f], acc[1][f]);
    }
    __syncthreads();
  }

  // ---- epilogue: bias / relu / store f32 ----
  #pragma unroll
  for (int f = 0; f < 4; ++f) {
    int ncol = gn0 + wn0 + f * 16 + lrow;
    float bv = bias ? bias[ncol] : 0.0f;
    #pragma unroll
    for (int mi = 0; mi < 2; ++mi) {
      #pragma unroll
      for (int e = 0; e < 8; ++e) {
        int mrow = gm0 + wm0 + mi * 16 + e + lhi * 8;
        float v = acc[mi][f][e] + bv;
        if (relu) v = fmaxf(v, 0.0f);
        C[(size_t)mrow * N + ncol] = v;
      }
    }
  }
}

// ---------------------------------------------------------------------------
// Attention: one wave per (b, h, 16-query tile).
// scores = (Q Kt)/sqrt(D_MODEL) + mask*(-1e9); softmax; O = P V.
// maskMode 0: mask[b*LK + k]   (padding mask over keys)
// maskMode 1: mask[q*LK + k]   (look-ahead mask)
// ---------------------------------------------------------------------------
__global__ __launch_bounds__(32)
void attn_wmma(const float* __restrict__ Q, const float* __restrict__ Kb,
               const float* __restrict__ Vb, const float* __restrict__ mask,
               int maskMode, float* __restrict__ O, int LQ, int LK) {
  __shared__ alignas(16) float          sc[16][SEQ];
  __shared__ alignas(16) unsigned short pb[16][SEQ];

  const int lane = threadIdx.x;
  const int qt = blockIdx.x, h = blockIdx.y, b = blockIdx.z;
  const int lrow = lane & 15, lhi = lane >> 4;
  const float scale = 0.03125f;   // 1/sqrt(D_MODEL)

  // Q fragments (d = 0..31, 32..63); runs of 8 floats are contiguous -> b128 loads
  FragAB aq[2];
  {
    const float* qp = Q + (size_t)(b * LQ + qt * 16 + lrow) * D_MODEL + h * DHEAD;
    #pragma unroll
    for (int f = 0; f < 2; ++f) {
      const float4* r0 = (const float4*)(qp + f * 32 + 8 * lhi);
      const float4* r1 = (const float4*)(qp + f * 32 + 8 * lhi + 16);
      pack8(aq[f].s,     r0[0], r0[1]);
      pack8(aq[f].s + 8, r1[0], r1[1]);
    }
  }

  // scores, tile-by-tile over keys (K-row fragment is two runs of 8 contiguous floats)
  for (int kt = 0; kt < LK / 16; ++kt) {
    int key = kt * 16 + lrow;
    const float* kp = Kb + (size_t)(b * LK + key) * D_MODEL + h * DHEAD;
    FragAB bk[2];
    #pragma unroll
    for (int f = 0; f < 2; ++f) {
      const float4* r = (const float4*)(kp + f * 32 + 8 * lhi);
      const float4* r1 = (const float4*)(kp + f * 32 + 8 * lhi + 16);
      pack8(bk[f].s,     r[0], r[1]);
      pack8(bk[f].s + 8, r1[0], r1[1]);
    }
    v8f a = {};
    a = wmma_bf16(aq[0], bk[0], a);
    a = wmma_bf16(aq[1], bk[1], a);
    #pragma unroll
    for (int e = 0; e < 8; ++e) {
      int m = e + 8 * lhi;
      int kc = kt * 16 + lrow;
      float mv = (maskMode == 0) ? mask[b * LK + kc]
                                 : mask[(qt * 16 + m) * LK + kc];
      sc[m][kc] = a[e] * scale + mv * (-1e9f);
    }
  }
  __syncthreads();

  // row softmax -> bf16 probabilities in LDS
  if (lane < 16) {
    float mx = -3.4e38f;
    for (int k2 = 0; k2 < LK; ++k2) mx = fmaxf(mx, sc[lane][k2]);
    float sum = 0.0f;
    for (int k2 = 0; k2 < LK; ++k2) {
      float e2 = __expf(sc[lane][k2] - mx);
      sc[lane][k2] = e2;
      sum += e2;
    }
    float inv = 1.0f / sum;
    for (int k2 = 0; k2 < LK; ++k2) pb[lane][k2] = f2bf_bits(sc[lane][k2] * inv);
  }
  __syncthreads();

  // O = P @ V
  v8f acc[4] = {};
  for (int ks = 0; ks < LK / 32; ++ks) {
    FragAB ap;
    const unsigned short* pr = &pb[lrow][ks * 32 + 16 * lhi];
    ap.u[0] = *(const u16x8*)(pr);
    ap.u[1] = *(const u16x8*)(pr + 8);
    #pragma unroll
    for (int f = 0; f < 4; ++f) {
      FragAB bv;
      int dcol = h * DHEAD + f * 16 + lrow;
      #pragma unroll
      for (int e = 0; e < 16; ++e) {
        int key = ks * 32 + 16 * lhi + e;
        bv.s[e] = f2bf_bits(Vb[(size_t)(b * LK + key) * D_MODEL + dcol]);
      }
      acc[f] = wmma_bf16(ap, bv, acc[f]);
    }
  }
  #pragma unroll
  for (int f = 0; f < 4; ++f)
    #pragma unroll
    for (int e = 0; e < 8; ++e) {
      int m = e + 8 * lhi;
      O[(size_t)(b * LQ + qt * 16 + m) * D_MODEL + h * DHEAD + f * 16 + lrow] = acc[f][e];
    }
}

// ---------------------------------------------------------------------------
// Y = LayerNorm(Aadd + X) * g + b  (eps = 1e-4). One block per row of 1024.
// Safe when Y aliases X.
// ---------------------------------------------------------------------------
__global__ __launch_bounds__(256)
void add_layernorm(const float* __restrict__ Aadd, const float* __restrict__ X,
                   const float* __restrict__ g, const float* __restrict__ be,
                   float* __restrict__ Y) {
  __shared__ float red[256];
  int row = blockIdx.x, tid = threadIdx.x;
  const float* a = Aadd + (size_t)row * D_MODEL;
  const float* x = X + (size_t)row * D_MODEL;
  float v[4], s = 0.0f;
  #pragma unroll
  for (int i = 0; i < 4; ++i) { v[i] = a[tid + i * 256] + x[tid + i * 256]; s += v[i]; }
  red[tid] = s; __syncthreads();
  for (int o = 128; o > 0; o >>= 1) { if (tid < o) red[tid] += red[tid + o]; __syncthreads(); }
  float mu = red[0] * (1.0f / D_MODEL);
  __syncthreads();
  float vs = 0.0f;
  #pragma unroll
  for (int i = 0; i < 4; ++i) { float d = v[i] - mu; vs += d * d; }
  red[tid] = vs; __syncthreads();
  for (int o = 128; o > 0; o >>= 1) { if (tid < o) red[tid] += red[tid + o]; __syncthreads(); }
  float rinv = rsqrtf(red[0] * (1.0f / D_MODEL) + 1e-4f);
  #pragma unroll
  for (int i = 0; i < 4; ++i) {
    int c = tid + i * 256;
    Y[(size_t)row * D_MODEL + c] = (v[i] - mu) * rinv * g[c] + be[c];
  }
}

// ---------------------------------------------------------------------------
// In-place row softmax over N columns. One block per row.
// ---------------------------------------------------------------------------
__global__ __launch_bounds__(256)
void softmax_rows(float* __restrict__ Z, int N) {
  __shared__ float red[256];
  int row = blockIdx.x, tid = threadIdx.x;
  float* z = Z + (size_t)row * N;
  float mx = -3.4e38f;
  for (int c = tid; c < N; c += 256) mx = fmaxf(mx, z[c]);
  red[tid] = mx; __syncthreads();
  for (int o = 128; o > 0; o >>= 1) { if (tid < o) red[tid] = fmaxf(red[tid], red[tid + o]); __syncthreads(); }
  mx = red[0]; __syncthreads();
  float s = 0.0f;
  for (int c = tid; c < N; c += 256) s += __expf(z[c] - mx);
  red[tid] = s; __syncthreads();
  for (int o = 128; o > 0; o >>= 1) { if (tid < o) red[tid] += red[tid + o]; __syncthreads(); }
  float inv = 1.0f / red[0];
  for (int c = tid; c < N; c += 256) z[c] = __expf(z[c] - mx) * inv;
}

// ---------------------------------------------------------------------------
extern "C" void kernel_launch(void* const* d_in, const int* in_sizes, int n_in,
                              void* d_out, int out_size, void* d_ws, size_t ws_size,
                              hipStream_t stream) {
  (void)in_sizes; (void)n_in; (void)out_size; (void)ws_size;
  const int*   toks_e  = (const int*)d_in[0];
  const int*   toks_d  = (const int*)d_in[1];
  const float* enc_msk = (const float*)d_in[2];
  const float* dec_msk = (const float*)d_in[3];
  const float* la_msk  = (const float*)d_in[4];
  const float* enc_emb = (const float*)d_in[5];
  const float* dec_emb = (const float*)d_in[6];
  auto P = [&](int i) { return (const float*)d_in[i]; };
  const size_t DD = (size_t)D_MODEL * D_MODEL;
  const size_t DF = (size_t)D_MODEL * D_FF;

  float* ws  = (float*)d_ws;
  const size_t SZ = (size_t)ROWS * D_MODEL;   // 2M floats
  float* x   = ws;            // encoder activations -> enc_out
  float* y   = ws + 1 * SZ;   // decoder activations
  float* t1  = ws + 2 * SZ;   // pre-LN temp
  float* qb  = ws + 3 * SZ;
  float* kb  = ws + 4 * SZ;
  float* vb  = ws + 5 * SZ;
  float* atn = ws + 6 * SZ;
  float* ffh = qb;            // FFN hidden (8M floats) aliases q/k/v/atn region

  auto gemm = [&](const float* A, const float* W, const float* bias, float* C,
                  int M, int N, int K, int relu) {
    dim3 g(N / TN, M / TM);
    gemm_bf16_wmma<<<g, 256, 0, stream>>>(A, W, bias, C, M, N, K, relu);
  };
  auto attn = [&](const float* q, const float* k, const float* v,
                  const float* m, int mode, float* o) {
    attn_wmma<<<dim3(SEQ / 16, N_HEAD, BATCH), 32, 0, stream>>>(q, k, v, m, mode, o, SEQ, SEQ);
  };
  auto lnorm = [&](const float* a, const float* xi, const float* g,
                   const float* b, float* yo) {
    add_layernorm<<<ROWS, 256, 0, stream>>>(a, xi, g, b, yo);
  };

  // ---------------- encoder ----------------
  embed_pe<<<ROWS * D_MODEL / 256, 256, 0, stream>>>(toks_e, enc_emb, x);
  for (int i = 0; i < 2; ++i) {
    gemm(x, P(7) + i * DD, nullptr, qb, ROWS, D_MODEL, D_MODEL, 0);   // wq
    gemm(x, P(8) + i * DD, nullptr, kb, ROWS, D_MODEL, D_MODEL, 0);   // wk
    gemm(x, P(9) + i * DD, nullptr, vb, ROWS, D_MODEL, D_MODEL, 0);   // wv
    attn(qb, kb, vb, enc_msk, 0, atn);
    gemm(atn, P(10) + i * DD, P(11) + i * D_MODEL, t1, ROWS, D_MODEL, D_MODEL, 0); // wo
    lnorm(t1, x, P(12) + i * D_MODEL, P(13) + i * D_MODEL, x);
    gemm(x, P(16) + i * DF, P(17) + i * D_FF, ffh, ROWS, D_FF, D_MODEL, 1);        // ffn1+relu
    gemm(ffh, P(18) + i * DF, P(19) + i * D_MODEL, t1, ROWS, D_MODEL, D_FF, 0);    // ffn2
    lnorm(t1, x, P(14) + i * D_MODEL, P(15) + i * D_MODEL, x);
  }
  // x == enc_out

  // ---------------- decoder ----------------
  embed_pe<<<ROWS * D_MODEL / 256, 256, 0, stream>>>(toks_d, dec_emb, y);
  for (int i = 0; i < 2; ++i) {
    // masked self-attention
    gemm(y, P(20) + i * DD, nullptr, qb, ROWS, D_MODEL, D_MODEL, 0);
    gemm(y, P(21) + i * DD, nullptr, kb, ROWS, D_MODEL, D_MODEL, 0);
    gemm(y, P(22) + i * DD, nullptr, vb, ROWS, D_MODEL, D_MODEL, 0);
    attn(qb, kb, vb, la_msk, 1, atn);
    gemm(atn, P(23) + i * DD, P(24) + i * D_MODEL, t1, ROWS, D_MODEL, D_MODEL, 0);
    lnorm(t1, y, P(30) + i * D_MODEL, P(31) + i * D_MODEL, y);
    // cross-attention over encoder output
    gemm(y, P(25) + i * DD, nullptr, qb, ROWS, D_MODEL, D_MODEL, 0);
    gemm(x, P(26) + i * DD, nullptr, kb, ROWS, D_MODEL, D_MODEL, 0);
    gemm(x, P(27) + i * DD, nullptr, vb, ROWS, D_MODEL, D_MODEL, 0);
    attn(qb, kb, vb, dec_msk, 0, atn);
    gemm(atn, P(28) + i * DD, P(29) + i * D_MODEL, t1, ROWS, D_MODEL, D_MODEL, 0);
    lnorm(t1, y, P(32) + i * D_MODEL, P(33) + i * D_MODEL, y);
    // FFN
    gemm(y, P(36) + i * DF, P(37) + i * D_FF, ffh, ROWS, D_FF, D_MODEL, 1);
    gemm(ffh, P(38) + i * DF, P(39) + i * D_MODEL, t1, ROWS, D_MODEL, D_FF, 0);
    lnorm(t1, y, P(34) + i * D_MODEL, P(35) + i * D_MODEL, y);
  }

  // ---------------- vocab projection + softmax ----------------
  float* out = (float*)d_out;
  gemm(y, P(40), P(41), out, ROWS, VOCAB, D_MODEL, 0);
  softmax_rows<<<ROWS, 256, 0, stream>>>(out, VOCAB);
}
